// Decoder_LSTM_38250978738660
// MI455X (gfx1250) — matrix operand order repaired and checked by reading, not compiled, not run on hardware
//
#include <hip/hip_runtime.h>

typedef __attribute__((ext_vector_type(16))) __bf16 v16bf;
typedef __attribute__((ext_vector_type(8)))  float  v8f;

#define F_DIM 64
#define G_DIM 256          // 4*F
#define WPAD  72           // padded LDS row (bf16 elems): 144B rows, 16B-aligned frags
#define T_STEPS 12
#define WAVES_PER_BLOCK 8
#define TOK_PER_WAVE 16
#define TOK_PER_BLOCK (WAVES_PER_BLOCK * TOK_PER_WAVE)

__device__ __forceinline__ v8f wmma_bf16(v16bf a, v16bf b, v8f c) {
  // D = A(16x32 bf16) * B(32x16 bf16) + C(16x16 f32)
  return __builtin_amdgcn_wmma_f32_16x16x32_bf16(
      /*neg_a=*/false, a, /*neg_b=*/false, b,
      /*c_mod=*/(short)0, c, /*reuse_a=*/false, /*reuse_b=*/false);
}

// CDNA5 V_TANH_F32 (TRANS pipe, co-executes with XDL WMMA); confirmed present
// (round 2 valu drop). Guard kept for portability.
#if __has_builtin(__builtin_amdgcn_tanhf)
__device__ __forceinline__ float fast_tanh(float x) {
  return __builtin_amdgcn_tanhf(x);
}
// Sigmoid of pre-halved argument: caller guarantees g already equals 0.5*x
// (0.5 folded into staged weights / Xg accumulators) -> just v_tanh + v_fma.
__device__ __forceinline__ float sigmoid_prehalved(float g) {
  return __builtin_fmaf(0.5f, __builtin_amdgcn_tanhf(g), 0.5f);
}
#else
__device__ __forceinline__ float fast_tanh(float x) {
  return 2.0f * __builtin_amdgcn_rcpf(1.0f + __expf(-2.0f * x)) - 1.0f;
}
__device__ __forceinline__ float sigmoid_prehalved(float g) {
  return __builtin_amdgcn_rcpf(1.0f + __expf(-2.0f * g));
}
#endif

// Load one 32x16 bf16 B-fragment from LDS: lane l holds column n=l%16,
// K = (l/16)*16 .. +15 contiguous -> 32 contiguous bytes per lane.
__device__ __forceinline__ v16bf lds_bfrag(const __bf16* p) {
  union { uint4 q[2]; v16bf v; } u;
  u.q[0] = *(const uint4*)(p);
  u.q[1] = *(const uint4*)(p + 8);
  return u.v;
}

__global__ __launch_bounds__(256, 1)
void lstm_wmma_bf16_kernel(const float* __restrict__ x,
                           const float* __restrict__ gate_w,   // (256,128) row-major
                           const float* __restrict__ gate_b,   // (256,)
                           const float* __restrict__ w_edge,   // (64,64) row-major
                           float* __restrict__ out,            // (T, ntok, 64)
                           int ntok)
{
  __shared__ __align__(16) __bf16 s_wh[G_DIM * WPAD];                    // Wh^T  [n][k]
  __shared__ __align__(16) __bf16 s_we[F_DIM * WPAD];                    // We^T  [n][k]
  __shared__ float s_b[G_DIM];
  __shared__ __align__(16) __bf16 s_h[WAVES_PER_BLOCK][TOK_PER_WAVE * F_DIM]; // h scratch

  const int tid  = threadIdx.x;
  const int lane = tid & 31;
  const int wave = tid >> 5;
  const int m    = lane & 15;   // A-row (token) / B-C-D column (feature-within-tile)
  const int half = lane >> 4;

  // Sigmoid gates (i: rows 0-63, f: 64-127, o: 192-255) carry a 0.5 pre-scale so
  // sigmoid(x) = 0.5*tanh(0.5x)+0.5 needs no per-element scaling in the loop.
  // For the h-path the 0.5 lives in the staged Wh rows; for the x-path + bias it
  // is applied once to the finished Xg accumulator (compile-time j, no selects).
  // Cell gate (rows 128-191) feeds plain tanh -> unscaled everywhere.
  auto gate_scale = [](int n) { return (n < 128 || n >= 192) ? 0.5f : 1.0f; };

  // ---- stage weights to LDS (bf16) ----
  for (int i = tid; i < G_DIM * F_DIM; i += 256) {
    int n = i >> 6, k = i & 63;
    s_wh[n * WPAD + k] = (__bf16)(gate_w[n * 128 + 64 + k] * gate_scale(n)); // B[k][n]=Wh[n][k]
  }
  for (int i = tid; i < F_DIM * F_DIM; i += 256) {
    int n = i >> 6, k = i & 63;
    s_we[n * WPAD + k] = (__bf16)(w_edge[k * 64 + n] * 0.5f);  // output sigmoid pre-halved
  }
  s_b[tid] = gate_b[tid];    // unscaled: Xg accumulator scale covers the bias
  __syncthreads();

  const int tokBase = (blockIdx.x * WAVES_PER_BLOCK + wave) * TOK_PER_WAVE;
  if (tokBase >= ntok) return;   // wave-uniform: EXEC stays all-ones below

  // K-position (within 32-wide K tile) of packed bf16 pair q of the A fragment
  auto kposA = [&](int q) { return q * 2 + half * 16; };

  // ---- build x A-fragments (16x32 bf16, contiguous k = 16*half + s) ----
  v16bf xa[2];
  const float* xrow = x + (size_t)(tokBase + m) * F_DIM;
  #pragma unroll
  for (int kt = 0; kt < 2; ++kt) {
    union { unsigned int w[8]; v16bf v; } u;
    #pragma unroll
    for (int q = 0; q < 8; ++q) {
      int k = kt * 32 + kposA(q);
      float2 f = *(const float2*)(xrow + k);
      union { unsigned int u32; __bf16 h[2]; } pk;
      pk.h[0] = (__bf16)f.x; pk.h[1] = (__bf16)f.y;
      u.w[q] = pk.u32;
    }
    xa[kt] = u.v;
  }

  // Wx B-fragment straight from global (one-time use, L2-resident), unscaled:
  // clean b128-load -> v_cvt_pk_bf16_f32 path.
  auto load_wx = [&](int j, int kt) {
    const float* p = gate_w + (j * 16 + m) * 128 + kt * 32 + half * 16;
    union { unsigned int w[8]; v16bf v; } u;
    #pragma unroll
    for (int q = 0; q < 8; ++q) {
      float2 f = *(const float2*)(p + 2 * q);
      union { unsigned int u32; __bf16 h[2]; } pk;
      pk.h[0] = (__bf16)f.x; pk.h[1] = (__bf16)f.y;
      u.w[q] = pk.u32;
    }
    return u.v;
  };

  // ---- Xg = scale_j * (x @ Wx^T + b) : time-invariant, kept in VGPRs ----
  v8f xg[16];
  #pragma unroll
  for (int j = 0; j < 16; ++j) {
    float bb = s_b[j * 16 + m];
    v8f acc;
    #pragma unroll
    for (int r = 0; r < 8; ++r) acc[r] = bb;
    acc = wmma_bf16(xa[0], load_wx(j, 0), acc);
    acc = wmma_bf16(xa[1], load_wx(j, 1), acc);
    if (j < 8 || j >= 12) {              // sigmoid tiles: fold the 0.5 here (j is compile-time)
      #pragma unroll
      for (int r = 0; r < 8; ++r) acc[r] *= 0.5f;
    }
    xg[j] = acc;
  }

  // ---- recurrent state: c in f32 regs (C/D layout), h as bf16 A-fragments ----
  v8f cst[4];
  #pragma unroll
  for (int ft = 0; ft < 4; ++ft) {
    #pragma unroll
    for (int r = 0; r < 8; ++r) cst[ft][r] = 0.0f;
  }
  v16bf ha[2];
  {
    union { unsigned int w[8]; v16bf v; } z;
    #pragma unroll
    for (int q = 0; q < 8; ++q) z.w[q] = 0u;
    ha[0] = z.v; ha[1] = z.v;
  }

  __bf16* hbuf = s_h[wave];   // per-wave private region: no barriers needed

  for (int t = 0; t < T_STEPS; ++t) {
    // --- gates = Xg + h @ Wh^T, then elementwise LSTM, 4 feature tiles ---
    #pragma unroll
    for (int ft = 0; ft < 4; ++ft) {
      v8f gi = xg[ft], gf = xg[ft + 4], gg = xg[ft + 8], go = xg[ft + 12];
      #pragma unroll
      for (int kt = 0; kt < 2; ++kt) {
        v16bf a = ha[kt];
        gi = wmma_bf16(a, lds_bfrag(s_wh + ((ft      ) * 16 + m) * WPAD + kt * 32 + half * 16), gi);
        gf = wmma_bf16(a, lds_bfrag(s_wh + ((ft +  4) * 16 + m) * WPAD + kt * 32 + half * 16), gf);
        gg = wmma_bf16(a, lds_bfrag(s_wh + ((ft +  8) * 16 + m) * WPAD + kt * 32 + half * 16), gg);
        go = wmma_bf16(a, lds_bfrag(s_wh + ((ft + 12) * 16 + m) * WPAD + kt * 32 + half * 16), go);
      }
      #pragma unroll
      for (int r = 0; r < 8; ++r) {
        float ig = sigmoid_prehalved(gi[r]);   // 0.5 carried by Xg scale + Wh rows
        float fg = sigmoid_prehalved(gf[r]);
        float cg = fast_tanh(gg[r]);
        float og = sigmoid_prehalved(go[r]);
        float cc = fg * cst[ft][r] + ig * cg;
        cst[ft][r] = cc;
        float hy = og * fast_tanh(cc);
        // C/D layout -> [token][feature] bf16 scratch (transpose via LDS)
        hbuf[(r + 8 * half) * F_DIM + ft * 16 + m] = (__bf16)hy;
      }
    }

    // --- regather h as A fragments for next matmuls ---
    #pragma unroll
    for (int kt = 0; kt < 2; ++kt) {
      union { unsigned int w[8]; v16bf v; } u;
      #pragma unroll
      for (int q = 0; q < 8; ++q) {
        int k = kt * 32 + kposA(q);
        u.w[q] = *(const unsigned int*)(hbuf + m * F_DIM + k);  // 2 bf16, k even
      }
      ha[kt] = u.v;
    }

    // --- y_t = sigmoid(h @ W_edge); W_edge pre-halved; NT stores (write-once) ---
    const size_t tbase = (size_t)t * (size_t)ntok * F_DIM + (size_t)tokBase * F_DIM;
    #pragma unroll
    for (int jo = 0; jo < 4; ++jo) {
      v8f ya;
      #pragma unroll
      for (int r = 0; r < 8; ++r) ya[r] = 0.0f;
      ya = wmma_bf16(ha[0], lds_bfrag(s_we + (jo * 16 + m) * WPAD +      half * 16), ya);
      ya = wmma_bf16(ha[1], lds_bfrag(s_we + (jo * 16 + m) * WPAD + 32 + half * 16), ya);
      #pragma unroll
      for (int r = 0; r < 8; ++r) {
        // lanes 0..15 write consecutive features of token (r+8*half): coalesced 64B runs
        __builtin_nontemporal_store(sigmoid_prehalved(ya[r]),
                                    &out[tbase + (size_t)(r + 8 * half) * F_DIM + jo * 16 + m]);
      }
    }
  }
}

extern "C" void kernel_launch(void* const* d_in, const int* in_sizes, int n_in,
                              void* d_out, int out_size, void* d_ws, size_t ws_size,
                              hipStream_t stream) {
  const float* x  = (const float*)d_in[0];   // (B,N,F) f32
  const float* gw = (const float*)d_in[1];   // (4F,2F) f32
  const float* gb = (const float*)d_in[2];   // (4F,)   f32
  const float* we = (const float*)d_in[3];   // (F,F)   f32
  float* out = (float*)d_out;                // (T,B,N,F) f32

  const int ntok = in_sizes[0] / F_DIM;      // B*N = 160000
  const int blocks = (ntok + TOK_PER_BLOCK - 1) / TOK_PER_BLOCK;
  lstm_wmma_bf16_kernel<<<blocks, 256, 0, stream>>>(x, gw, gb, we, out, ntok);
}
